// FAPAT_72129680769673
// MI455X (gfx1250) — matrix operand order repaired and checked by reading, not compiled, not run on hardware
//
#include <hip/hip_runtime.h>

// ---------------- types ----------------
typedef __attribute__((ext_vector_type(16))) __bf16 v16bf;
typedef __attribute__((ext_vector_type(8)))  float  v8f;

#define KK 4
#define BB 64
#define SS 50
#define DD 256
#define HH 4
#define DH 64
#define VV 40000
#define BS (BB*SS)          // 3200
#define NEGBIG (-1e9f)

// ---------------- scalar helpers ----------------
__device__ __forceinline__ unsigned short f2bf(float f) {
    union { float f; unsigned int u; } x; x.f = f;
    unsigned int r = x.u + 0x7FFFu + ((x.u >> 16) & 1u);
    return (unsigned short)(r >> 16);
}
__device__ __forceinline__ float bf2f(unsigned short h) {
    union { unsigned int u; float f; } x; x.u = ((unsigned int)h) << 16;
    return x.f;
}
__device__ __forceinline__ float gelu_t(float x) {
    float x3 = x * x * x;
    float t = tanhf(0.7978845608028654f * (x + 0.044715f * x3));
    return 0.5f * x * (1.f + t);
}

// ---------------- WMMA fragment loaders ----------------
// A fragment (16x32 bf16, M x K): lane<16 -> row=lane, K={kb..kb+7, kb+16..kb+23}
//                                 lane>=16 -> row=lane-16, K={kb+8..15, kb+24..31}
__device__ __forceinline__ v16bf fragA(const unsigned short* p, int ld, int lane) {
    int r  = lane & 15;
    int ko = (lane & 16) ? 8 : 0;
    const unsigned short* q = p + (size_t)r * ld + ko;
    union { uint4 u[2]; v16bf v; } t;
    t.u[0] = *(const uint4*)(q);
    t.u[1] = *(const uint4*)(q + 16);
    return t.v;
}
// B fragment (32x16 bf16, K x N) read from BT[n][k] storage:
// lane<16 -> col n=lane, K={kb..kb+7, kb+16..kb+23} ; lane>=16 -> col n=lane-16, K={kb+8..15, kb+24..31}
__device__ __forceinline__ v16bf fragB(const unsigned short* p, int ld, int lane) {
    int n  = lane & 15;
    int ko = (lane & 16) ? 8 : 0;
    const unsigned short* q = p + (size_t)n * ld + ko;
    union { uint4 u[2]; v16bf v; } t;
    t.u[0] = *(const uint4*)(q);
    t.u[1] = *(const uint4*)(q + 16);
    return t.v;
}
__device__ __forceinline__ v8f wmma_bf16(v16bf a, v16bf b, v8f c) {
    return __builtin_amdgcn_wmma_f32_16x16x32_bf16(false, a, false, b, (short)0, c, false, false);
}

// ---------------- kernel: weight convert + transpose (f32 [K][R][C] -> bf16 [K][C][R]) ----
__global__ __launch_bounds__(256)
void wconv(const float* __restrict__ W, unsigned short* __restrict__ WT,
           int R, int C, int total) {
    int idx = blockIdx.x * 256 + threadIdx.x;
    if (idx >= total) return;
    int rc = R * C;
    int k = idx / rc;
    int rem = idx - k * rc;
    int r = rem / C;
    int c = rem - r * C;
    WT[(size_t)k * rc + (size_t)c * R + r] = f2bf(W[idx]);
}

// ---------------- kernel: embedding gathers (hg bf16) + h_seq mean (f32) ----------------
__global__ __launch_bounds__(256)
void gather_k(const int* __restrict__ ids, const int* __restrict__ items,
              const float* __restrict__ emb,
              unsigned short* __restrict__ hgbf, float* __restrict__ hseq) {
    int bs = blockIdx.x;       // b*S+s
    int d  = threadIdx.x;
    float acc = 0.f;
    for (int k = 0; k < KK; k++) {
        int it = items[k * BS + bs];
        int id = ids[k * BS + bs];
        hgbf[((size_t)k * BS + bs) * DD + d] = f2bf(emb[((size_t)k * VV + it) * DD + d]);
        acc += emb[((size_t)k * VV + id) * DD + d];
    }
    hseq[(size_t)bs * DD + d] = acc * 0.25f;
}

// ---------------- kernel: local graph aggregator per (k,b), ~224KB dynamic LDS ----------
// LDS map (bytes):
//   hgb    [64][256] bf16   @      0  (32768)
//   hgbT   [256][64] bf16   @  32768  (32768)
//   hgr    [4][64][256] bf16@  65536  (131072)  -- reused as hl [64][256] f32 after softmax
//   alphaS [64][64] f32     @ 196608  (16384)
//   palpha [64][64] bf16    @ 212992  (8192)
//   adjc   [64][64] u8      @ 221184  (4096)
//   aw     [4][256] f32     @ 225280  (4096)
//   aliasS [64] i32         @ 229376  (256)
#define LAGG_SMEM (229376 + 256)
__global__ __launch_bounds__(256)
void local_agg(const unsigned short* __restrict__ hgbf,
               const float* __restrict__ agg_a,
               const int* __restrict__ adj, const int* __restrict__ alias,
               float* __restrict__ h) {
    extern __shared__ char smem[];
    unsigned short* hgb    = (unsigned short*)smem;
    unsigned short* hgbT   = (unsigned short*)(smem + 32768);
    unsigned short* hgr    = (unsigned short*)(smem + 65536);
    float*          hl     = (float*)(smem + 65536);           // alias of hgr
    float*          alphaS = (float*)(smem + 196608);
    unsigned short* palpha = (unsigned short*)(smem + 212992);
    unsigned char*  adjc   = (unsigned char*)(smem + 221184);
    float*          aw     = (float*)(smem + 225280);
    int*            aliasS = (int*)(smem + 229376);

    int b = blockIdx.x & 63, k = blockIdx.x >> 6;
    int tid = threadIdx.x, lane = tid & 31, w = tid >> 5;

    // stage inputs
    for (int i = tid; i < 64 * 256; i += 256) { hgb[i] = 0; hgbT[i] = 0; }
    for (int i = tid; i < 4 * 256;  i += 256) aw[i] = agg_a[(size_t)k * 4 * DD + i];
    for (int i = tid; i < 64 * 64;  i += 256) adjc[i] = 0;
    if (tid < SS) aliasS[tid] = alias[b * SS + tid];
    __syncthreads();

    const unsigned short* hg = hgbf + ((size_t)k * BS + b * SS) * DD;
    for (int i = tid; i < SS * DD; i += 256) {
        int s = i >> 8, d = i & 255;
        unsigned short hv = hg[i];
        hgb[s * 256 + d] = hv;
        hgbT[d * 64 + s] = hv;
    }
    const int* adjb = adj + (size_t)b * SS * SS;
    for (int i = tid; i < SS * SS; i += 256) {
        int r = i / SS, c = i - r * SS;
        adjc[r * 64 + c] = (unsigned char)adjb[i];
    }
    __syncthreads();

    // precompute hgr[r][j][d] = bf16(hg[j][d] * a[r][d]); zero pad rows >= 50
    for (int i = tid; i < 4 * (64 - SS) * 256; i += 256) {
        int r = i / ((64 - SS) * 256);
        int rem = i - r * (64 - SS) * 256;
        hgr[r * 64 * 256 + SS * 256 + rem] = 0;
    }
    for (int i = tid; i < 4 * SS * 256; i += 256) {
        int r = i / (SS * 256);
        int rem = i - r * SS * 256;
        int d = rem & 255;
        hgr[r * 64 * 256 + rem] = f2bf(bf2f(hgb[rem]) * aw[r * 256 + d]);
    }
    __syncthreads();

    // e[r,i,j] = hg @ hgr[r]^T : 4 accumulators per tile, branch-free select epilogue
    for (int t = w; t < 16; t += 8) {
        int mi = t >> 2, nj = t & 3;
        v8f acc[4];
#pragma unroll
        for (int r = 0; r < 4; r++) acc[r] = (v8f){0.f,0.f,0.f,0.f,0.f,0.f,0.f,0.f};
        for (int kb = 0; kb < 8; kb++) {
            v16bf a = fragA(hgb + mi * 16 * 256 + kb * 32, 256, lane);
#pragma unroll
            for (int r = 0; r < 4; r++) {
                v16bf bb = fragB(hgr + r * 64 * 256 + nj * 16 * 256 + kb * 32, 256, lane);
                acc[r] = wmma_bf16(a, bb, acc[r]);
            }
        }
        int lr = lane >> 4, lc = lane & 15;
#pragma unroll
        for (int v = 0; v < 8; v++) {
            int i = mi * 16 + v + 8 * lr, j = nj * 16 + lc;
            int av = (int)adjc[i * 64 + j];
            float sel = (av == 1) ? acc[0][v]
                      : (av == 2) ? acc[1][v]
                      : (av == 3) ? acc[2][v]
                      : (av == 4) ? acc[3][v] : 0.f;
            float lk  = sel > 0.f ? sel : 0.2f * sel;
            alphaS[i * 64 + j] = (av >= 1 && av <= 4) ? lk : NEGBIG;
        }
    }
    __syncthreads();

    // row softmax over j<50, store bf16 probs (0-padded to 64)
    if (tid < 64) {
        float mx = -3.4e38f;
        for (int j = 0; j < SS; j++) mx = fmaxf(mx, alphaS[tid * 64 + j]);
        float sum = 0.f;
        for (int j = 0; j < SS; j++) sum += expf(alphaS[tid * 64 + j] - mx);
        float inv = 1.f / sum;
        for (int j = 0; j < 64; j++) {
            float p = (j < SS) ? expf(alphaS[tid * 64 + j] - mx) * inv : 0.f;
            palpha[tid * 64 + j] = f2bf(p);
        }
    }
    __syncthreads();   // also fences last reads of hgr before hl overwrites it

    // h_local = alpha @ hg   (A: palpha, BT: hgbT) -> hl (aliases hgr space)
    for (int t = w * 8; t < w * 8 + 8; t++) {
        int mi = t >> 4, nd = t & 15;
        v8f acc = {0.f,0.f,0.f,0.f,0.f,0.f,0.f,0.f};
        for (int kb = 0; kb < 2; kb++) {
            v16bf a  = fragA(palpha + mi * 16 * 64 + kb * 32, 64, lane);
            v16bf bb = fragB(hgbT   + nd * 16 * 64 + kb * 32, 64, lane);
            acc = wmma_bf16(a, bb, acc);
        }
        int lr = lane >> 4, lc = lane & 15;
#pragma unroll
        for (int v = 0; v < 8; v++)
            hl[(mi * 16 + v + 8 * lr) * 256 + nd * 16 + lc] = acc[v];
    }
    __syncthreads();

    // alias scatter: h[k,b,i,:] = hl[alias[b,i],:]
    float* hrow = h + ((size_t)k * BS + b * SS) * DD;
    for (int i = 0; i < SS; i++) {
        int al = aliasS[i];
        hrow[i * DD + tid] = hl[al * 256 + tid];
    }
}

// ---------------- kernel: LayerNorm (f32 in -> bf16 out) ----------------
__global__ __launch_bounds__(256)
void ln_bf16(const float* __restrict__ in, const float* __restrict__ g,
             const float* __restrict__ bta, unsigned short* __restrict__ out) {
    int row = blockIdx.x;
    int k = row / BS;
    int d = threadIdx.x;
    __shared__ float r1[256], r2[256];
    const float* x = in + (size_t)row * DD;
    float v = x[d];
    r1[d] = v; r2[d] = v * v;
    __syncthreads();
    for (int s = 128; s > 0; s >>= 1) {
        if (d < s) { r1[d] += r1[d + s]; r2[d] += r2[d + s]; }
        __syncthreads();
    }
    float m = r1[0] * (1.f / DD);
    float var = r2[0] * (1.f / DD) - m * m;
    float y = (v - m) * rsqrtf(var + 1e-5f) * g[k * DD + d] + bta[k * DD + d];
    out[(size_t)row * DD + d] = f2bf(y);
}

// ---------------- kernel: batched bf16 WMMA GEMM, 64x64 tile / block ----------------
// C[k] = A[k] (MxKd bf16) x BT[k] (NxKd bf16, i.e. B transposed)
// Compile-time epilogue: ACT(0 none,1 gelu), RESID, OUTF, OUTB.
template <int ACT, bool RESID, bool OUTF, bool OUTB>
__global__ __launch_bounds__(256)
void gemm_bf16(const unsigned short* __restrict__ A,
               const unsigned short* __restrict__ BT,
               float* __restrict__ outF, unsigned short* __restrict__ outB,
               const float* __restrict__ resid,
               int M, int N, int Kd) {
    int k = blockIdx.z;
    const unsigned short* Ak = A  + (size_t)k * M * Kd;
    const unsigned short* Bk = BT + (size_t)k * N * Kd;
    size_t obase = (size_t)k * M * N;
    int bm = blockIdx.y * 64, bn = blockIdx.x * 64;
    int w = threadIdx.x >> 5, lane = threadIdx.x & 31;
    int mi = w & 3, nj = (w >> 2) * 32;

    v8f acc0 = {0.f,0.f,0.f,0.f,0.f,0.f,0.f,0.f};
    v8f acc1 = {0.f,0.f,0.f,0.f,0.f,0.f,0.f,0.f};
    const unsigned short* arow  = Ak + (size_t)(bm + mi * 16) * Kd;
    const unsigned short* b0row = Bk + (size_t)(bn + nj) * Kd;
    const unsigned short* b1row = Bk + (size_t)(bn + nj + 16) * Kd;
    for (int kb = 0; kb < Kd; kb += 32) {
        if (kb + 32 < Kd) {   // prefetch next k-slab (global_prefetch_b8)
            __builtin_prefetch(arow  + kb + 32, 0, 0);
            __builtin_prefetch(b0row + kb + 32, 0, 0);
            __builtin_prefetch(b1row + kb + 32, 0, 0);
        }
        v16bf a  = fragA(arow  + kb, Kd, lane);
        v16bf b0 = fragB(b0row + kb, Kd, lane);
        v16bf b1 = fragB(b1row + kb, Kd, lane);
        acc0 = wmma_bf16(a, b0, acc0);
        acc1 = wmma_bf16(a, b1, acc1);
    }
    int lr = lane >> 4, lc = lane & 15;
    int n0 = bn + nj + lc, n1 = n0 + 16;
#pragma unroll
    for (int v = 0; v < 8; v++) {
        int m = bm + mi * 16 + v + 8 * lr;
        float f0 = acc0[v], f1 = acc1[v];
        if (RESID) {
            f0 += resid[obase + (size_t)m * N + n0];
            f1 += resid[obase + (size_t)m * N + n1];
        }
        if (ACT == 1) { f0 = gelu_t(f0); f1 = gelu_t(f1); }
        if (OUTF) {
            outF[obase + (size_t)m * N + n0] = f0;
            outF[obase + (size_t)m * N + n1] = f1;
        }
        if (OUTB) {
            outB[obase + (size_t)m * N + n0] = f2bf(f0);
            outB[obase + (size_t)m * N + n1] = f2bf(f1);
        }
    }
}

// ---------------- kernel: causal MHA per (k,b,h), 48KB static LDS ----------------
__global__ __launch_bounds__(256)
void attn_k(const unsigned short* __restrict__ qb,
            const unsigned short* __restrict__ kbuf,
            const unsigned short* __restrict__ vb,
            unsigned short* __restrict__ ao) {
    __shared__ __attribute__((aligned(16))) unsigned short qs[64 * 64];
    __shared__ __attribute__((aligned(16))) unsigned short ks[64 * 64];
    __shared__ __attribute__((aligned(16))) unsigned short vts[64 * 64];
    __shared__ __attribute__((aligned(16))) unsigned short ps[64 * 64];
    __shared__ __attribute__((aligned(16))) float sc[64 * 64];

    int h = blockIdx.x, b = blockIdx.y, k = blockIdx.z;
    int tid = threadIdx.x, lane = tid & 31, w = tid >> 5;
    size_t base = ((size_t)k * BS + b * SS) * DD + h * DH;

    for (int i = tid; i < 4096; i += 256) {
        int s = i >> 6, c = i & 63;
        unsigned short qv = 0, kv = 0, vv = 0;
        if (s < SS) {
            size_t off = base + (size_t)s * DD + c;
            qv = qb[off]; kv = kbuf[off]; vv = vb[off];
        }
        qs[i] = qv; ks[i] = kv; vts[c * 64 + s] = vv;
    }
    __syncthreads();

    // scores = Q K^T
    for (int t = w; t < 16; t += 8) {
        int mi = t >> 2, nj = t & 3;
        v8f acc = {0.f,0.f,0.f,0.f,0.f,0.f,0.f,0.f};
        for (int kb = 0; kb < 2; kb++) {
            v16bf a  = fragA(qs + mi * 16 * 64 + kb * 32, 64, lane);
            v16bf bb = fragB(ks + nj * 16 * 64 + kb * 32, 64, lane);
            acc = wmma_bf16(a, bb, acc);
        }
        int lr = lane >> 4, lc = lane & 15;
#pragma unroll
        for (int v = 0; v < 8; v++)
            sc[(mi * 16 + v + 8 * lr) * 64 + nj * 16 + lc] = acc[v];
    }
    __syncthreads();

    // causal scale + softmax per row
    if (tid < 64) {
        int i = tid;
        float mx = -3.4e38f;
        for (int j = 0; j < SS; j++) {
            float sv = (j <= i) ? sc[i * 64 + j] * 0.125f : NEGBIG;
            sc[i * 64 + j] = sv;
            mx = fmaxf(mx, sv);
        }
        float sum = 0.f;
        for (int j = 0; j < SS; j++) sum += expf(sc[i * 64 + j] - mx);
        float inv = 1.f / sum;
        for (int j = 0; j < 64; j++) {
            float p = (j < SS) ? expf(sc[i * 64 + j] - mx) * inv : 0.f;
            ps[i * 64 + j] = f2bf(p);
        }
    }
    __syncthreads();

    // O = P V   (BT operand = V^T in vts)
    for (int t = w; t < 16; t += 8) {
        int mi = t >> 2, nc = t & 3;
        v8f acc = {0.f,0.f,0.f,0.f,0.f,0.f,0.f,0.f};
        for (int kb = 0; kb < 2; kb++) {
            v16bf a  = fragA(ps  + mi * 16 * 64 + kb * 32, 64, lane);
            v16bf bb = fragB(vts + nc * 16 * 64 + kb * 32, 64, lane);
            acc = wmma_bf16(a, bb, acc);
        }
        int lr = lane >> 4, lc = lane & 15;
#pragma unroll
        for (int v = 0; v < 8; v++) {
            int s = mi * 16 + v + 8 * lr;
            if (s < SS)
                ao[base + (size_t)s * DD + nc * 16 + lc] = f2bf(acc[v]);
        }
    }
}

// ---------------- kernel: noisy top-2 gating + expert mix ----------------
__global__ __launch_bounds__(256)
void gate_k(const float* __restrict__ hseq, const float* __restrict__ Wg,
            const float* __restrict__ Wn, const float* __restrict__ eps,
            const float* __restrict__ hkeys, float* __restrict__ out) {
    int bs = blockIdx.x;
    int d  = threadIdx.x;
    __shared__ float red[256];
    __shared__ float logits[KK];
    __shared__ float gates[KK];
    float hv = hseq[(size_t)bs * DD + d];
    for (int kk = 0; kk < KK; kk++) {
        red[d] = hv * Wg[d * KK + kk];
        __syncthreads();
        for (int s = 128; s > 0; s >>= 1) { if (d < s) red[d] += red[d + s]; __syncthreads(); }
        float g = red[0];
        __syncthreads();
        red[d] = hv * Wn[d * KK + kk];
        __syncthreads();
        for (int s = 128; s > 0; s >>= 1) { if (d < s) red[d] += red[d + s]; __syncthreads(); }
        if (d == 0) {
            float n = red[0];
            float sp = (n > 20.f) ? n : log1pf(expf(n));
            logits[kk] = g + sp * eps[(size_t)bs * KK + kk];
        }
        __syncthreads();
    }
    if (d == 0) {
        int i1 = 0;
        for (int kk = 1; kk < KK; kk++) if (logits[kk] > logits[i1]) i1 = kk;
        int i2 = -1;
        for (int kk = 0; kk < KK; kk++) {
            if (kk == i1) continue;
            if (i2 < 0 || logits[kk] > logits[i2]) i2 = kk;
        }
        float e2 = expf(logits[i2] - logits[i1]);
        float w1 = 1.f / (1.f + e2), w2 = e2 / (1.f + e2);
        gates[0] = gates[1] = gates[2] = gates[3] = 0.f;
        gates[i1] = w1; gates[i2] = w2;
    }
    __syncthreads();
    float acc = 0.f;
    for (int kk = 0; kk < KK; kk++)
        acc += gates[kk] * hkeys[((size_t)kk * BS + bs) * DD + d];
    out[(size_t)bs * DD + d] = acc;
}

// ---------------- host launcher ----------------
extern "C" void kernel_launch(void* const* d_in, const int* in_sizes, int n_in,
                              void* d_out, int out_size, void* d_ws, size_t ws_size,
                              hipStream_t stream) {
    (void)in_sizes; (void)n_in; (void)out_size; (void)ws_size;
    const int*   ids    = (const int*)d_in[0];
    const int*   items  = (const int*)d_in[1];
    const int*   adj    = (const int*)d_in[2];
    const int*   alias  = (const int*)d_in[3];
    const float* eps    = (const float*)d_in[4];
    const float* emb    = (const float*)d_in[5];
    const float* agg_a  = (const float*)d_in[6];
    const float* Wq     = (const float*)d_in[7];
    const float* Wk     = (const float*)d_in[8];
    const float* Wv     = (const float*)d_in[9];
    const float* Wo     = (const float*)d_in[10];
    const float* ln1g   = (const float*)d_in[11];
    const float* ln1b   = (const float*)d_in[12];
    const float* W1     = (const float*)d_in[13];
    const float* W2     = (const float*)d_in[14];
    const float* ln2g   = (const float*)d_in[15];
    const float* ln2b   = (const float*)d_in[16];
    const float* Wg     = (const float*)d_in[17];
    const float* Wn     = (const float*)d_in[18];
    float* out = (float*)d_out;

    // workspace carve
    char* ws = (char*)d_ws;
    size_t off = 0;
    auto alloc = [&](size_t bytes) -> char* {
        char* p = ws + off;
        off += (bytes + 255) & ~(size_t)255;
        return p;
    };
    unsigned short* WqT  = (unsigned short*)alloc((size_t)KK * DD * DD * 2);
    unsigned short* WkT  = (unsigned short*)alloc((size_t)KK * DD * DD * 2);
    unsigned short* WvT  = (unsigned short*)alloc((size_t)KK * DD * DD * 2);
    unsigned short* WoT  = (unsigned short*)alloc((size_t)KK * DD * DD * 2);
    unsigned short* W1T  = (unsigned short*)alloc((size_t)KK * DD * 4 * DD * 2);
    unsigned short* W2T  = (unsigned short*)alloc((size_t)KK * 4 * DD * DD * 2);
    unsigned short* hgbf = (unsigned short*)alloc((size_t)KK * BS * DD * 2);
    float*          hseq = (float*)alloc((size_t)BS * DD * 4);
    float*          h    = (float*)alloc((size_t)KK * BS * DD * 4);
    unsigned short* x_bf = (unsigned short*)alloc((size_t)KK * BS * DD * 2);
    unsigned short* q_bf = (unsigned short*)alloc((size_t)KK * BS * DD * 2);
    unsigned short* k_bf = (unsigned short*)alloc((size_t)KK * BS * DD * 2);
    unsigned short* v_bf = (unsigned short*)alloc((size_t)KK * BS * DD * 2);
    unsigned short* ao_bf= (unsigned short*)alloc((size_t)KK * BS * DD * 2);
    unsigned short* u_bf = (unsigned short*)alloc((size_t)KK * BS * 4 * DD * 2);

    // 1) weight conversion (transposed bf16)
    {
        int tq = KK * DD * DD;        // 262144
        wconv<<<(tq + 255) / 256, 256, 0, stream>>>(Wq, WqT, DD, DD, tq);
        wconv<<<(tq + 255) / 256, 256, 0, stream>>>(Wk, WkT, DD, DD, tq);
        wconv<<<(tq + 255) / 256, 256, 0, stream>>>(Wv, WvT, DD, DD, tq);
        wconv<<<(tq + 255) / 256, 256, 0, stream>>>(Wo, WoT, DD, DD, tq);
        int t1 = KK * DD * 4 * DD;    // 1048576
        wconv<<<(t1 + 255) / 256, 256, 0, stream>>>(W1, W1T, DD, 4 * DD, t1);
        wconv<<<(t1 + 255) / 256, 256, 0, stream>>>(W2, W2T, 4 * DD, DD, t1);
    }

    // 2) gathers + gate-input mean
    gather_k<<<BS, 256, 0, stream>>>(ids, items, emb, hgbf, hseq);

    // 3) local graph aggregator (~224KB dynamic LDS: exploit 320KB/WGP)
    {
        (void)hipFuncSetAttribute((const void*)local_agg,
                                  hipFuncAttributeMaxDynamicSharedMemorySize, LAGG_SMEM);
        local_agg<<<KK * BB, 256, LAGG_SMEM, stream>>>(hgbf, agg_a, adj, alias, h);
    }

    // 4) LN1 -> x_bf
    ln_bf16<<<KK * BS, 256, 0, stream>>>(h, ln1g, ln1b, x_bf);

    // 5) Q/K/V projections (bf16 out)
    {
        dim3 g(DD / 64, BS / 64, KK);
        gemm_bf16<0, false, false, true><<<g, 256, 0, stream>>>(x_bf, WqT, nullptr, q_bf, nullptr, BS, DD, DD);
        gemm_bf16<0, false, false, true><<<g, 256, 0, stream>>>(x_bf, WkT, nullptr, k_bf, nullptr, BS, DD, DD);
        gemm_bf16<0, false, false, true><<<g, 256, 0, stream>>>(x_bf, WvT, nullptr, v_bf, nullptr, BS, DD, DD);
    }

    // 6) causal attention per (k,b,h)
    {
        dim3 g(HH, BB, KK);
        attn_k<<<g, 256, 0, stream>>>(q_bf, k_bf, v_bf, ao_bf);
    }

    // 7) output projection + residual (in-place into h)
    {
        dim3 g(DD / 64, BS / 64, KK);
        gemm_bf16<0, true, true, false><<<g, 256, 0, stream>>>(ao_bf, WoT, h, nullptr, h, BS, DD, DD);
    }

    // 8) LN2 -> x_bf (reused)
    ln_bf16<<<KK * BS, 256, 0, stream>>>(h, ln2g, ln2b, x_bf);

    // 9) FFN1 + gelu (bf16 out)
    {
        dim3 g(4 * DD / 64, BS / 64, KK);
        gemm_bf16<1, false, false, true><<<g, 256, 0, stream>>>(x_bf, W1T, nullptr, u_bf, nullptr, BS, 4 * DD, DD);
    }

    // 10) FFN2 + residual (in-place into h -> final h_keys)
    {
        dim3 g(DD / 64, BS / 64, KK);
        gemm_bf16<0, true, true, false><<<g, 256, 0, stream>>>(u_bf, W2T, h, nullptr, h, BS, DD, 4 * DD);
    }

    // 11) noisy top-2 gating + expert mix -> out
    gate_k<<<BS, 256, 0, stream>>>(hseq, Wg, Wn, eps, h, out);
}
